// AttentionBlock_80487687127051
// MI455X (gfx1250) — matrix operand order repaired and checked by reading, not compiled, not run on hardware
//
#include <hip/hip_runtime.h>
#include <hip/hip_bf16.h>

// ---------------------------------------------------------------------------
// AttentionBlock for MI455X (gfx1250): bf16 WMMA 16x16x32, f32 accumulate.
// keys = X@Wk+bk ; logits = keys@keys^T/32 (causal, masked=-100) ; softmax ;
// read = P@values ; out = concat(X, read).
// ---------------------------------------------------------------------------

typedef __bf16 bf16_t;
typedef __attribute__((ext_vector_type(16))) __bf16 v16bf;
typedef __attribute__((ext_vector_type(8)))  __bf16 v8bf;
typedef __attribute__((ext_vector_type(8)))  float  v8f;

#define DIMS   1024
#define S_LEN  1000
#define S_PAD  1008      // 63 tiles of 16
#define NB     16
#define PS_LD  1032      // LDS row stride (bf16): 2064B -> rows land on distinct banks

__device__ __forceinline__ v8f vzero8() {
    v8f z = {0.f, 0.f, 0.f, 0.f, 0.f, 0.f, 0.f, 0.f};
    return z;
}

__device__ __forceinline__ v8f wmma_bf16(v16bf a, v16bf b, v8f c) {
    // D = A(16x32) x B(32x16) + C(16x16 f32)
    return __builtin_amdgcn_wmma_f32_16x16x32_bf16(
        /*neg_a=*/false, a, /*neg_b=*/false, b,
        /*c_mod=*/(short)0, c, /*reuse_a=*/false, /*reuse_b=*/false);
}

// A-fragment (16x32 bf16, row-major source, row0 = M tile origin):
// Contiguous k order used here (B matches): lanes 0-15 -> M=lane,  K=k0..k0+15
//             lanes16-31 -> M=lane-16, K=k0+16..k0+31
__device__ __forceinline__ v16bf load_frag_a(const bf16_t* __restrict__ base,
                                             int ld, int row0, int k0) {
    int lane = threadIdx.x & 31;
    int r  = row0 + (lane & 15);
    int kk = k0 + ((lane >> 4) << 4);           // +16 for upper lane half
    const bf16_t* p = base + (long)r * ld + kk; // 16B-aligned chunks
    v8bf lo = *(const v8bf*)(p);
    v8bf hi = *(const v8bf*)(p + 8);
    v16bf a;
#pragma unroll
    for (int i = 0; i < 8; ++i) { a[i] = lo[i]; a[i + 8] = hi[i]; }
    return a;
}

// B-fragment (32x16 bf16) where B[k][n] = srcT[n0+n][k0+k] (srcT row-major):
// Contiguous k order, same as A: lanes 0-15 -> N=lane,  K=k0..k0+15 ; lanes 16-31 -> N, K=k0+16..+31
// => one contiguous 32-byte load per lane.
__device__ __forceinline__ v16bf load_frag_b(const bf16_t* __restrict__ srcT,
                                             int ld, int n0, int k0) {
    int lane = threadIdx.x & 31;
    int n  = n0 + (lane & 15);
    int kk = k0 + ((lane >> 4) << 4);           // +16 for upper lane half
    return *(const v16bf*)(srcT + (long)n * ld + kk);
}

// ---------------------------------------------------------------------------
// 1) fp32 -> bf16 X (zero-padded rows s>=1000), and copy X into out[:, 0:1024]
//    grid: (S_PAD*DIMS/256, NB) -> all index math is shifts/masks
// ---------------------------------------------------------------------------
__global__ void convert_kernel(const float* __restrict__ x,
                               bf16_t* __restrict__ Xb,
                               float* __restrict__ out) {
    int idx = blockIdx.x * 256 + threadIdx.x;   // < S_PAD*DIMS
    int b   = blockIdx.y;
    int s   = idx >> 10;                        // DIMS == 1024
    int d   = idx & 1023;
    long xi = ((long)b * S_PAD + s) * DIMS + d;
    if (s < S_LEN) {
        float v = x[((long)b * S_LEN + s) * DIMS + d];
        Xb[xi] = (bf16_t)v;
        out[((long)b * S_LEN + s) * (2 * DIMS) + d] = v;
    } else {
        Xb[xi] = (bf16_t)0.f;
    }
}

// ---------------------------------------------------------------------------
// 2) W [in][out] fp32 -> Wt [out][in] bf16 (both Wk and Wv)
// ---------------------------------------------------------------------------
__global__ void transw_kernel(const float* __restrict__ Wk,
                              const float* __restrict__ Wv,
                              bf16_t* __restrict__ Wkt,
                              bf16_t* __restrict__ Wvt) {
    int i = blockIdx.x * blockDim.x + threadIdx.x;
    if (i >= DIMS * DIMS) return;
    int o = i >> 10, in = i & 1023;
    Wkt[i] = (bf16_t)Wk[in * DIMS + o];
    Wvt[i] = (bf16_t)Wv[in * DIMS + o];
}

// ---------------------------------------------------------------------------
// 3) keys = X@Wk+bk (bf16, [b][s_pad][k]); values^T = (X@Wv+bv)^T ([b][v][s_pad])
//    grid: (63 s-tiles, 8 col-groups, NB batches); block = 8 waves;
//    wave w -> 16x16 tile of keys AND values (A reused); 64 WMMAs/wave
// ---------------------------------------------------------------------------
__global__ __launch_bounds__(256) void proj_kernel(
    const bf16_t* __restrict__ Xb, const bf16_t* __restrict__ Wkt,
    const bf16_t* __restrict__ Wvt, const float* __restrict__ bk,
    const float* __restrict__ bv, bf16_t* __restrict__ keys,
    bf16_t* __restrict__ valsT) {
    int wave = threadIdx.x >> 5, lane = threadIdx.x & 31;
    int b  = blockIdx.z;
    int s0 = blockIdx.x * 16;
    int m0 = b * S_PAD + s0;                    // flat padded row
    int n0 = blockIdx.y * 128 + wave * 16;
    v8f accK = vzero8(), accV = vzero8();
#pragma unroll 4
    for (int k0 = 0; k0 < DIMS; k0 += 32) {
        v16bf a  = load_frag_a(Xb, DIMS, m0, k0);
        v16bf fk = load_frag_b(Wkt, DIMS, n0, k0);
        accK = wmma_bf16(a, fk, accK);
        v16bf fv = load_frag_b(Wvt, DIMS, n0, k0);
        accV = wmma_bf16(a, fv, accV);
    }
    int col   = n0 + (lane & 15);
    int rbase = (lane < 16) ? 0 : 8;
    float bkv = bk[col], bvv = bv[col];
    bf16_t* krow = keys  + (long)(m0 + rbase) * DIMS + col;
    bf16_t* vrow = valsT + ((long)b * DIMS + col) * S_PAD + s0 + rbase;
#pragma unroll
    for (int r = 0; r < 8; ++r) {
        krow[(long)r * DIMS] = (bf16_t)(accK[r] + bkv);
        vrow[r]              = (bf16_t)(accV[r] + bvv);
    }
}

// ---------------------------------------------------------------------------
// 4) Flash-style causal attention, one block per (q-tile, batch).
//    Pass 1: online row max/sum over s-tiles (waves round-robin).
//    Pass 2: exp-probs -> LDS (bf16). Then P@V, wave owns 128 V columns.
// ---------------------------------------------------------------------------
__global__ __launch_bounds__(256) void attn_kernel(
    const bf16_t* __restrict__ keys, const bf16_t* __restrict__ valsT,
    float* __restrict__ out) {
    __shared__ __align__(32) bf16_t Ps[16 * PS_LD];
    __shared__ float Mred[8][16];
    __shared__ float Lred[8][16];

    int wave = threadIdx.x >> 5, lane = threadIdx.x & 31;
    int qt = blockIdx.x, b = blockIdx.y;
    int q0 = qt * 16;
    const bf16_t* kb = keys  + (long)b * S_PAD * DIMS;
    const bf16_t* vb = valsT + (long)b * DIMS * S_PAD;
    int rbase = (lane < 16) ? 0 : 8;
    int colL  = lane & 15;

    // zero the probs buffer (covers the odd-tile K padding for P@V)
    for (int i = threadIdx.x; i < (16 * PS_LD) / 2; i += 256)
        ((unsigned int*)Ps)[i] = 0u;

    // ---- pass 1: online softmax statistics -------------------------------
    float m[8], l[8];
#pragma unroll
    for (int r = 0; r < 8; ++r) { m[r] = -1e30f; l[r] = 0.f; }

    for (int st = wave; st <= qt; st += 8) {
        v8f acc = vzero8();
        for (int k0 = 0; k0 < DIMS; k0 += 32) {
            v16bf a  = load_frag_a(kb, DIMS, q0, k0);
            v16bf bb = load_frag_b(kb, DIMS, st * 16, k0);
            acc = wmma_bf16(a, bb, acc);
        }
        int sg = st * 16 + colL;
#pragma unroll
        for (int r = 0; r < 8; ++r) {
            float v = acc[r] * 0.03125f;            // 1/sqrt(1024)
            int qg = q0 + rbase + r;
            if (sg > qg || sg >= S_LEN) v = -100.f; // reference mask value
            float tm = v;                            // row max over 16 lanes
            for (int mk = 1; mk < 16; mk <<= 1) tm = fmaxf(tm, __shfl_xor(tm, mk, 32));
            float nm = fmaxf(m[r], tm);
            float e  = __expf(v - nm);
            for (int mk = 1; mk < 16; mk <<= 1) e += __shfl_xor(e, mk, 32);
            l[r] = l[r] * __expf(m[r] - nm) + e;
            m[r] = nm;
        }
    }
    if (lane == 0 || lane == 16) {
#pragma unroll
        for (int r = 0; r < 8; ++r) { Mred[wave][rbase + r] = m[r]; Lred[wave][rbase + r] = l[r]; }
    }
    __syncthreads();

    float mt[8], lt[8];
#pragma unroll
    for (int r = 0; r < 8; ++r) {
        int row = rbase + r;
        float mm = -1e30f;
        for (int w = 0; w < 8; ++w) mm = fmaxf(mm, Mred[w][row]);
        float ll = 0.f;
        for (int w = 0; w < 8; ++w) ll += Lred[w][row] * __expf(Mred[w][row] - mm);
        mt[r] = mm; lt[r] = ll;
    }
    __syncthreads();

    // ---- pass 2: probs -> LDS --------------------------------------------
    for (int st = wave; st <= qt; st += 8) {
        v8f acc = vzero8();
        for (int k0 = 0; k0 < DIMS; k0 += 32) {
            v16bf a  = load_frag_a(kb, DIMS, q0, k0);
            v16bf bb = load_frag_b(kb, DIMS, st * 16, k0);
            acc = wmma_bf16(a, bb, acc);
        }
        int sg = st * 16 + colL;
        bf16_t* prow = Ps + rbase * PS_LD + sg;
#pragma unroll
        for (int r = 0; r < 8; ++r) {
            float v = acc[r] * 0.03125f;
            int qg = q0 + rbase + r;
            if (sg > qg || sg >= S_LEN) v = -100.f;
            prow[r * PS_LD] = (bf16_t)__expf(v - mt[r]);
        }
    }
    __syncthreads();

    // ---- P @ V : wave owns v-columns [wave*128, wave*128+128) ------------
    int send = (((qt + 1) * 16) + 31) & ~31;     // pad K loop to 32 (Ps zeroed)
    v8f accv[8];
#pragma unroll
    for (int t = 0; t < 8; ++t) accv[t] = vzero8();
    for (int k0 = 0; k0 < send; k0 += 32) {
        v16bf a = load_frag_a(Ps, PS_LD, 0, k0);
#pragma unroll
        for (int t = 0; t < 8; ++t) {
            v16bf bb = load_frag_b(vb, S_PAD, wave * 128 + t * 16, k0);
            accv[t] = wmma_bf16(a, bb, accv[t]);
        }
    }
    float* obase = out + ((long)(b * S_LEN + q0 + rbase)) * (2 * DIMS) + DIMS
                 + wave * 128 + colL;
#pragma unroll
    for (int r = 0; r < 8; ++r) {
        int q = q0 + rbase + r;
        if (q < S_LEN) {
            float inv = 1.f / lt[r];
#pragma unroll
            for (int t = 0; t < 8; ++t)
                obase[(long)r * (2 * DIMS) + t * 16] = accv[t][r] * inv;
        }
    }
}

// ---------------------------------------------------------------------------
extern "C" void kernel_launch(void* const* d_in, const int* in_sizes, int n_in,
                              void* d_out, int out_size, void* d_ws, size_t ws_size,
                              hipStream_t stream) {
    const float* x  = (const float*)d_in[0];   // [16,1000,1024]
    const float* Wk = (const float*)d_in[1];   // [1024,1024] (in,out)
    const float* bk = (const float*)d_in[2];   // [1024]
    const float* Wv = (const float*)d_in[3];   // [1024,1024]
    const float* bv = (const float*)d_in[4];   // [1024]
    // d_in[5] = seq_len (==1000), compile-time constant here
    float* out = (float*)d_out;

    char* ws = (char*)d_ws;
    size_t off = 0;
    auto take = [&](size_t bytes) {
        void* p = ws + off;
        off += (bytes + 255) & ~(size_t)255;
        return p;
    };
    bf16_t* Xb    = (bf16_t*)take((size_t)NB * S_PAD * DIMS * 2);
    bf16_t* keysb = (bf16_t*)take((size_t)NB * S_PAD * DIMS * 2);
    bf16_t* valsT = (bf16_t*)take((size_t)NB * DIMS * S_PAD * 2);
    bf16_t* Wkt   = (bf16_t*)take((size_t)DIMS * DIMS * 2);
    bf16_t* Wvt   = (bf16_t*)take((size_t)DIMS * DIMS * 2);

    convert_kernel<<<dim3(S_PAD * DIMS / 256, NB), dim3(256), 0, stream>>>(x, Xb, out);
    transw_kernel<<<dim3((DIMS * DIMS + 255) / 256), dim3(256), 0, stream>>>(Wk, Wv, Wkt, Wvt);
    proj_kernel<<<dim3(S_PAD / 16, DIMS / 128, NB), dim3(256), 0, stream>>>(
        Xb, Wkt, Wvt, bk, bv, keysb, valsT);
    attn_kernel<<<dim3(S_PAD / 16, NB), dim3(256), 0, stream>>>(keysb, valsT, out);
}
